// Transformer_28278064677264
// MI455X (gfx1250) — matrix operand order repaired and checked by reading, not compiled, not run on hardware
//
#include <hip/hip_runtime.h>
#include <hip/hip_bf16.h>
#include <math.h>
#include <stdint.h>

// ---------------------------------------------------------------------------
// CDNA5 (gfx1250) wave32 WMMA types
// ---------------------------------------------------------------------------
typedef __attribute__((ext_vector_type(16))) __bf16 bf16x16;
typedef __attribute__((ext_vector_type(8)))  __bf16 bf16x8;
typedef __attribute__((ext_vector_type(8)))  float  f32x8;

#if defined(__gfx1250__) && __has_builtin(__builtin_amdgcn_tensor_load_to_lds) && \
    __has_builtin(__builtin_amdgcn_s_wait_tensorcnt)
#define HAVE_TDM 1
#else
#define HAVE_TDM 0
#endif

__device__ __forceinline__ f32x8 wmma_bf16(bf16x16 a, bf16x16 b, f32x8 c) {
    // D = A(16x32 bf16) x B(32x16 bf16) + C(16x16 f32)
    return __builtin_amdgcn_wmma_f32_16x16x32_bf16(
        /*neg_a=*/false, a, /*neg_b=*/false, b,
        /*c_mod=*/(short)0, c, /*reuse_a=*/false, /*reuse_b=*/false);
}

// ---------------------------------------------------------------------------
// Fragment loaders (layouts per ISA 7.12.2, wave32).
// A 16x32 bf16: lane=(m=lane&15, half=lane>>4); elem e<8 -> K=half*8+e,
//               e>=8 -> K=16+half*8+(e-8)  => two contiguous 16B runs.
// B 32x16 bf16: lane n=lane&15 picks source row; elem e<8 -> K=half*8+e,
//               e>=8 -> K=16+half*8+(e-8)  => two contiguous 16B runs.
// C/D 16x16 f32: reg r -> M=r+8*half, N=lane&15.
// ---------------------------------------------------------------------------
__device__ __forceinline__ bf16x16 load_a_frag_bf16(const __bf16* __restrict__ A,
                                                    int lda, int lane) {
    int m = lane & 15, half = lane >> 4;
    const __bf16* base = A + (size_t)m * lda + (half << 3);
    bf16x8 lo = *(const bf16x8*)(base);        // K = half*8 + 0..7
    bf16x8 hi = *(const bf16x8*)(base + 16);   // K = 16 + half*8 + 0..7
    return __builtin_shufflevector(lo, hi, 0, 1, 2, 3, 4, 5, 6, 7,
                                   8, 9, 10, 11, 12, 13, 14, 15);
}

__device__ __forceinline__ bf16x16 load_b_frag_bf16(const __bf16* __restrict__ W,
                                                    int ldw, int lane) {
    int n = lane & 15, half = lane >> 4;
    const __bf16* base = W + (size_t)n * ldw + (half << 3);
    bf16x8 lo = *(const bf16x8*)(base);        // K = half*8 + 0..7
    bf16x8 hi = *(const bf16x8*)(base + 16);   // K = 16 + half*8 + 0..7
    return __builtin_shufflevector(lo, hi, 0, 1, 2, 3, 4, 5, 6, 7,
                                   8, 9, 10, 11, 12, 13, 14, 15);
}

// A-fragment from an fp32 16x32 tile in LDS (P relayout in attention)
__device__ __forceinline__ bf16x16 load_a_frag_lds_f32(const float* __restrict__ P,
                                                       int lane) {
    int m = lane & 15, half = lane >> 4;
    bf16x16 a;
#pragma unroll
    for (int i = 0; i < 8; ++i) {
        int k = ((i >> 2) << 4) + (half << 3) + ((i & 3) << 1);
        a[2 * i]     = (__bf16)P[m * 32 + k];
        a[2 * i + 1] = (__bf16)P[m * 32 + k + 1];
    }
    return a;
}

__device__ __forceinline__ float rowmax16(float v) {
    v = fmaxf(v, __shfl_xor(v, 1, 32));
    v = fmaxf(v, __shfl_xor(v, 2, 32));
    v = fmaxf(v, __shfl_xor(v, 4, 32));
    v = fmaxf(v, __shfl_xor(v, 8, 32));
    return v;
}
__device__ __forceinline__ float rowsum16(float v) {
    v += __shfl_xor(v, 1, 32);
    v += __shfl_xor(v, 2, 32);
    v += __shfl_xor(v, 4, 32);
    v += __shfl_xor(v, 8, 32);
    return v;
}

// ---------------------------------------------------------------------------
// Tensor Data Mover: 2D bf16 tile (rows x cols, row stride in elems) -> LDS.
// D# packing per CDNA5 ISA 8.3/8.4 (count=1, data_size=2B, type=2).
// This toolchain exposes the 6-arg builtin:
//   (uint32x4 g0, int32x8 g1, int32x4 g2, int32x4 g3, int32x8, i32 cpol)
// ---------------------------------------------------------------------------
#if HAVE_TDM
typedef __attribute__((ext_vector_type(4))) unsigned int u32x4;
typedef __attribute__((ext_vector_type(8))) int          i32x8;
typedef __attribute__((ext_vector_type(4))) int          i32x4;

__device__ __forceinline__ void tdm_load_2d_bf16(__bf16* lds_dst,
                                                 const __bf16* gsrc,
                                                 int rows, int cols,
                                                 int row_stride) {
    unsigned long long ga = (unsigned long long)(uintptr_t)gsrc;
    unsigned int lds_off  = (unsigned int)(uintptr_t)lds_dst;  // LDS byte offset
    u32x4 g0;
    g0[0] = 1u;                                            // count=1 (valid D#)
    g0[1] = lds_off;                                       // lds_addr
    g0[2] = (unsigned int)(ga & 0xFFFFFFFFu);              // global_addr lo
    g0[3] = (unsigned int)((ga >> 32) & 0x1FFFFFFu) | (2u << 30);  // hi | type=2
    i32x8 g1;
    g1[0] = 1 << 16;                                       // data_size=2B, mask=0
    g1[1] = (cols & 0xFFFF) << 16;                         // tensor_dim0 lo
    g1[2] = ((cols >> 16) & 0xFFFF) | ((rows & 0xFFFF) << 16);  // dim0 hi|dim1 lo
    g1[3] = ((rows >> 16) & 0xFFFF) | ((cols & 0xFFFF) << 16);  // dim1 hi|tile_dim0
    g1[4] = rows & 0xFFFF;                                 // tile_dim1 (tile_dim2=0)
    g1[5] = row_stride;                                    // tensor_dim0_stride lo32
    g1[6] = 0;
    g1[7] = 0;
    i32x4 z4 = {0, 0, 0, 0};
    i32x8 z8 = {0, 0, 0, 0, 0, 0, 0, 0};
    __builtin_amdgcn_tensor_load_to_lds(g0, g1, z4, z4, z8, 0);
}
#endif

// ---------------------------------------------------------------------------
// fp32 -> bf16 conversion (weights, once per launch). 4 elems/thread.
// ---------------------------------------------------------------------------
__global__ __launch_bounds__(256) void cvt_bf16_kernel(
        const float* __restrict__ src, __bf16* __restrict__ dst, int n4) {
    int i = (blockIdx.x * blockDim.x + threadIdx.x) * 4;
#pragma unroll
    for (int j = 0; j < 4; ++j) dst[i + j] = (__bf16)src[i + j];
}

// ---------------------------------------------------------------------------
// Embedding + sinusoidal positional encoding; f32 master + bf16 shadow.
// ---------------------------------------------------------------------------
__global__ __launch_bounds__(256) void embed_kernel(
        const float* __restrict__ src, const float* __restrict__ emb_w,
        const float* __restrict__ emb_b, float* __restrict__ x,
        __bf16* __restrict__ xb, int S) {
    int idx = blockIdx.x * blockDim.x + threadIdx.x;  // tok*512 + d
    int d   = idx & 511;
    int tok = idx >> 9;
    int s   = tok & (S - 1);
    const float* sp = src + (size_t)tok * 25;
    const float* wp = emb_w + (size_t)d * 25;
    float acc = emb_b[d];
#pragma unroll
    for (int i = 0; i < 25; ++i) acc += sp[i] * wp[i];
    int   j   = d >> 1;
    float dv  = __expf(-(float)(2 * j) * 0.017988946039015984f);  // ln(1e4)/512
    float ang = (float)s * dv;
    acc += (d & 1) ? cosf(ang) : sinf(ang);
    x[idx]  = acc;
    xb[idx] = (__bf16)acc;
}

// ---------------------------------------------------------------------------
// GEMM: C[M,N] = A[M,K](bf16) * W[N,K](bf16)^T + bias.
// Block = 8 waves sharing one 32-row A stripe (NT % 8 == 0 for every N used):
// the 32x512 A chunk is staged into LDS once per block (TDM if available),
// then each wave runs a 32x64 macro-tile: 8 WMMAs per 4 B-fragment loads.
// ---------------------------------------------------------------------------
template <int N, int K, bool RELU, bool OUTBF>
__global__ __launch_bounds__(256) void gemm_wmma(
        const __bf16* __restrict__ A, const __bf16* __restrict__ W,
        const float* __restrict__ bias, float* __restrict__ C,
        __bf16* __restrict__ Cb, int M) {
    constexpr int NT = N / 64;            // N macro-tiles, multiple of 8
    constexpr int CW = 512;               // K chunk width staged in LDS
    __shared__ __bf16 As[32 * CW];        // 32 KB

    int wib  = threadIdx.x >> 5;
    int lane = threadIdx.x & 31;
    int wid  = blockIdx.x * 8 + wib;
    int mt = wid / NT, nt = wid % NT;     // mt uniform across the block

    const __bf16* Ap = A + (size_t)mt * 32 * K;
    const __bf16* Wp = W + (size_t)nt * 64 * K;
    int n15 = lane & 15, half = lane >> 4;

    f32x8 c[2][4];
#pragma unroll
    for (int j = 0; j < 4; ++j) {
        float bv = bias[nt * 64 + j * 16 + n15];
#pragma unroll
        for (int i = 0; i < 2; ++i)
#pragma unroll
            for (int r = 0; r < 8; ++r) c[i][j][r] = bv;
    }

    for (int kc = 0; kc < K; kc += CW) {
        __syncthreads();                  // protect As reuse across chunks
#if HAVE_TDM
        if (wib == 0) {                   // one TDM descriptor per block
            tdm_load_2d_bf16(As, Ap + kc, 32, CW, K);
            __builtin_amdgcn_s_wait_tensorcnt(0);
        }
#else
        for (int t = threadIdx.x * 8; t < 32 * CW; t += 256 * 8) {
            int row = t / CW, col = t % CW;
            *(bf16x8*)(As + row * CW + col) =
                *(const bf16x8*)(Ap + (size_t)row * K + kc + col);
        }
#endif
        __syncthreads();

#pragma unroll 4
        for (int k0 = 0; k0 < CW; k0 += 32) {
            bf16x16 a0 = load_a_frag_bf16(As + k0, CW, lane);           // ds_load
            bf16x16 a1 = load_a_frag_bf16(As + 16 * CW + k0, CW, lane);
#pragma unroll
            for (int j = 0; j < 4; ++j) {
                bf16x16 b = load_b_frag_bf16(Wp + (size_t)(j * 16) * K + kc + k0,
                                             K, lane);
                c[0][j] = wmma_bf16(a0, b, c[0][j]);
                c[1][j] = wmma_bf16(a1, b, c[1][j]);
            }
        }
    }

#pragma unroll
    for (int i = 0; i < 2; ++i)
#pragma unroll
        for (int j = 0; j < 4; ++j)
#pragma unroll
            for (int r = 0; r < 8; ++r) {
                float v = c[i][j][r];
                if (RELU) v = fmaxf(v, 0.0f);
                size_t row = (size_t)(mt * 32 + i * 16 + r + 8 * half);
                size_t col = (size_t)(nt * 64 + j * 16 + n15);
                if (OUTBF) Cb[row * N + col] = (__bf16)v;
                else       C[row * N + col]  = v;
            }
}

// ---------------------------------------------------------------------------
// V transpose: Vt[(b*4+h)*128 + d][s] = qkvb[b*S+s][1024 + h*128 + d]
// Makes the P*V B-fragment two contiguous 16B loads per lane.
// ---------------------------------------------------------------------------
__global__ __launch_bounds__(256) void v_transpose_kernel(
        const __bf16* __restrict__ qkvb, __bf16* __restrict__ vt, int S) {
    int idx  = blockIdx.x * blockDim.x + threadIdx.x;  // ((bh*128+d)*S + s)
    int s    = idx & (S - 1);
    int rest = idx >> 11;                              // S = 2048
    int d    = rest & 127;
    int bh   = rest >> 7;
    int b = bh >> 2, h = bh & 3;
    vt[idx] = qkvb[(size_t)(b * S + s) * 1536 + 1024 + h * 128 + d];
}

// ---------------------------------------------------------------------------
// Flash-style causal attention (PRE=20 block unmasked); all-bf16 operands.
// qkvb: [B*S,1536] bf16, vt: [B*H*128, S] bf16, out: [B*S,512] bf16.
// ---------------------------------------------------------------------------
#define ATT_WPB 4
__global__ __launch_bounds__(ATT_WPB * 32) void attn_wmma(
        const __bf16* __restrict__ qkvb, const __bf16* __restrict__ vt,
        __bf16* __restrict__ out, int S) {
    int wib  = threadIdx.x >> 5;
    int lane = threadIdx.x & 31;
    int gw   = blockIdx.x * ATT_WPB + wib;
    int qTiles = S >> 4;
    int total  = 16 * qTiles;                 // B(4) * H(4) * qTiles
    if (gw >= total) return;                  // whole-wave uniform
    int qt = gw % qTiles;
    int bh = gw / qTiles;
    int h = bh & 3, b = bh >> 2;
    int q_base = qt << 4;

    const int ld = 1536;
    const __bf16* qp  = qkvb + (size_t)b * S * ld + h * 128;
    const __bf16* kp  = qp + 512;
    const __bf16* vtp = vt + (size_t)bh * 128 * S;

    int n15 = lane & 15, half = lane >> 4;

    __shared__ float Pl[ATT_WPB][16][32];
    float* P = &Pl[wib][0][0];

    bf16x16 qa[4];
#pragma unroll
    for (int kc = 0; kc < 4; ++kc)
        qa[kc] = load_a_frag_bf16(qp + (size_t)q_base * ld + kc * 32, ld, lane);

    float mi[8], li[8], cr[8];
    f32x8 acc[8];
#pragma unroll
    for (int r = 0; r < 8; ++r) { mi[r] = -1.0e30f; li[r] = 0.0f; }
#pragma unroll
    for (int t = 0; t < 8; ++t)
#pragma unroll
        for (int r = 0; r < 8; ++r) acc[t][r] = 0.0f;

    const float sc = 0.08838834764831845f;    // 1/sqrt(128)
    int nkb = (q_base + 16 + 31) >> 5;
    for (int kb = 0; kb < nkb; ++kb) {
        int key0 = kb << 5;
        f32x8 s0, s1;
#pragma unroll
        for (int r = 0; r < 8; ++r) { s0[r] = 0.0f; s1[r] = 0.0f; }
#pragma unroll
        for (int kc = 0; kc < 4; ++kc) {
            bf16x16 b0 = load_b_frag_bf16(kp + (size_t)key0 * ld + kc * 32, ld, lane);
            bf16x16 b1 = load_b_frag_bf16(kp + (size_t)(key0 + 16) * ld + kc * 32,
                                          ld, lane);
            s0 = wmma_bf16(qa[kc], b0, s0);
            s1 = wmma_bf16(qa[kc], b1, s1);
        }
        float p0[8], p1[8];
#pragma unroll
        for (int r = 0; r < 8; ++r) {
            int q  = q_base + r + 8 * half;
            int c0 = key0 + n15;
            int c1 = c0 + 16;
            float v0 = ((c0 <= q) || (q < 20 && c0 < 20)) ? s0[r] * sc : -1.0e30f;
            float v1 = ((c1 <= q) || (q < 20 && c1 < 20)) ? s1[r] * sc : -1.0e30f;
            float rm = rowmax16(fmaxf(v0, v1));
            float mn = fmaxf(mi[r], rm);
            float e0 = __expf(v0 - mn);
            float e1 = __expf(v1 - mn);
            float rs = rowsum16(e0 + e1);
            float co = __expf(mi[r] - mn);
            li[r] = li[r] * co + rs;
            mi[r] = mn;
            cr[r] = co;
            p0[r] = e0; p1[r] = e1;
        }
#pragma unroll
        for (int t = 0; t < 8; ++t)
#pragma unroll
            for (int r = 0; r < 8; ++r) acc[t][r] *= cr[r];

        // C-frag -> A-frag relayout through per-wave LDS
#pragma unroll
        for (int r = 0; r < 8; ++r) {
            int mm = r + 8 * half;
            P[mm * 32 + n15]      = p0[r];
            P[mm * 32 + 16 + n15] = p1[r];
        }
        asm volatile("s_wait_dscnt 0" ::: "memory");  // CDNA5 split DS counter
        bf16x16 pa = load_a_frag_lds_f32(P, lane);

#pragma unroll
        for (int t = 0; t < 8; ++t) {
            bf16x16 bv = load_b_frag_bf16(vtp + (size_t)(t * 16) * S + key0, S, lane);
            acc[t] = wmma_bf16(pa, bv, acc[t]);
        }
    }

    __bf16* op = out + (size_t)(b * S + q_base) * 512 + h * 128;
#pragma unroll
    for (int t = 0; t < 8; ++t)
#pragma unroll
        for (int r = 0; r < 8; ++r)
            op[(size_t)(r + 8 * half) * 512 + t * 16 + n15] =
                (__bf16)(acc[t][r] / li[r]);
}

// ---------------------------------------------------------------------------
// x = LayerNorm(x + res) * g + b; writes f32 master + bf16 shadow.
// ---------------------------------------------------------------------------
__global__ __launch_bounds__(256) void ln_residual_kernel(
        float* __restrict__ x, __bf16* __restrict__ xb,
        const float* __restrict__ res,
        const float* __restrict__ g, const float* __restrict__ bta) {
    int row = blockIdx.x;
    int tid = threadIdx.x;
    size_t base = (size_t)row * 512;
    float v0 = x[base + tid] + res[base + tid];
    float v1 = x[base + 256 + tid] + res[base + 256 + tid];
    float s  = v0 + v1;
    float ss = v0 * v0 + v1 * v1;
#pragma unroll
    for (int off = 16; off >= 1; off >>= 1) {
        s  += __shfl_xor(s, off, 32);
        ss += __shfl_xor(ss, off, 32);
    }
    __shared__ float sh[16];
    int w = tid >> 5, ln = tid & 31;
    if (ln == 0) { sh[w] = s; sh[8 + w] = ss; }
    __syncthreads();
    if (tid == 0) {
        float ts = 0.0f, tss = 0.0f;
#pragma unroll
        for (int i = 0; i < 8; ++i) { ts += sh[i]; tss += sh[8 + i]; }
        sh[0] = ts; sh[8] = tss;
    }
    __syncthreads();
    float mu   = sh[0] * (1.0f / 512.0f);
    float var  = sh[8] * (1.0f / 512.0f) - mu * mu;
    float rstd = rsqrtf(var + 1e-5f);
    float o0 = (v0 - mu) * rstd * g[tid] + bta[tid];
    float o1 = (v1 - mu) * rstd * g[tid + 256] + bta[tid + 256];
    x[base + tid]        = o0;
    x[base + 256 + tid]  = o1;
    xb[base + tid]       = (__bf16)o0;
    xb[base + 256 + tid] = (__bf16)o1;
}

// ---------------------------------------------------------------------------
// out[row] = tanh(dot(x[row], dec_w) + dec_b) * 10    (one wave per row)
// ---------------------------------------------------------------------------
__global__ __launch_bounds__(256) void decode_kernel(
        const float* __restrict__ x, const float* __restrict__ dec_w,
        const float* __restrict__ dec_b, float* __restrict__ out, int rows) {
    int wid  = (blockIdx.x * blockDim.x + threadIdx.x) >> 5;
    int lane = threadIdx.x & 31;
    if (wid >= rows) return;
    const float* xr = x + (size_t)wid * 512;
    float s = 0.0f;
#pragma unroll
    for (int i = 0; i < 16; ++i) s += xr[lane + 32 * i] * dec_w[lane + 32 * i];
#pragma unroll
    for (int off = 16; off >= 1; off >>= 1) s += __shfl_xor(s, off, 32);
    if (lane == 0) out[wid] = tanhf(s + dec_b[0]) * 10.0f;
}

// ---------------------------------------------------------------------------
// Host-side orchestration
// ---------------------------------------------------------------------------
extern "C" void kernel_launch(void* const* d_in, const int* in_sizes, int n_in,
                              void* d_out, int out_size, void* d_ws, size_t ws_size,
                              hipStream_t stream) {
    const float* src   = (const float*)d_in[0];
    const float* emb_w = (const float*)d_in[1];
    const float* emb_b = (const float*)d_in[2];
    const float* in_w  = (const float*)d_in[3];
    const float* in_b  = (const float*)d_in[4];
    const float* out_w = (const float*)d_in[5];
    const float* out_b = (const float*)d_in[6];
    const float* ln1_g = (const float*)d_in[7];
    const float* ln1_b = (const float*)d_in[8];
    const float* ln2_g = (const float*)d_in[9];
    const float* ln2_b = (const float*)d_in[10];
    const float* ff1_w = (const float*)d_in[11];
    const float* ff1_b = (const float*)d_in[12];
    const float* ff2_w = (const float*)d_in[13];
    const float* ff2_b = (const float*)d_in[14];
    const float* dec_w = (const float*)d_in[15];
    const float* dec_b = (const float*)d_in[16];
    float* out = (float*)d_out;

    char* ws = (char*)d_ws;
    float*  x     = (float*) (ws + ((size_t)0   << 20)); // 16 MB [8192,512] f32
    __bf16* xb    = (__bf16*)(ws + ((size_t)16  << 20)); //  8 MB bf16 shadow
    __bf16* qkvb  = (__bf16*)(ws + ((size_t)24  << 20)); // 24 MB [8192,1536] bf16
    __bf16* vt    = (__bf16*)(ws + ((size_t)48  << 20)); //  8 MB [16*128, S] bf16
    __bf16* hbf   = (__bf16*)(ws + ((size_t)56  << 20)); // 32 MB [8192,2048] bf16
    float*  f     = (float*) (ws + ((size_t)88  << 20)); // 16 MB [8192,512] f32
    __bf16* ob    = (__bf16*)(ws + ((size_t)104 << 20)); //  8 MB [8192,512] bf16
    __bf16* w_in  = (__bf16*)(ws + ((size_t)112 << 20)); // 4.5 MB 3x1536x512
    __bf16* w_out = (__bf16*)(ws + ((size_t)117 << 20)); // 1.5 MB 3x512x512
    __bf16* w_ff1 = (__bf16*)(ws + ((size_t)119 << 20)); //   6 MB 3x2048x512
    __bf16* w_ff2 = (__bf16*)(ws + ((size_t)125 << 20)); //   6 MB 3x512x2048

    const int S = 2048;
    const int M = 4 * S;  // 8192 tokens

    cvt_bf16_kernel<<<(3 * 1536 * 512) / 1024, 256, 0, stream>>>(in_w,  w_in,  0);
    cvt_bf16_kernel<<<(3 * 512 * 512)  / 1024, 256, 0, stream>>>(out_w, w_out, 0);
    cvt_bf16_kernel<<<(3 * 2048 * 512) / 1024, 256, 0, stream>>>(ff1_w, w_ff1, 0);
    cvt_bf16_kernel<<<(3 * 512 * 2048) / 1024, 256, 0, stream>>>(ff2_w, w_ff2, 0);

    embed_kernel<<<(M * 512) / 256, 256, 0, stream>>>(src, emb_w, emb_b, x, xb, S);

    for (int l = 0; l < 3; ++l) {
        // QKV: xb x w_in^T -> qkvb (bf16)
        gemm_wmma<1536, 512, false, true><<<(256 * 24) / 8, 256, 0, stream>>>(
            xb, w_in + (size_t)l * 1536 * 512, in_b + l * 1536, nullptr, qkvb, M);
        // V transpose for contiguous P*V fragments
        v_transpose_kernel<<<(16 * 128 * S) / 256, 256, 0, stream>>>(qkvb, vt, S);
        // attention -> ob (bf16)
        attn_wmma<<<(16 * (S >> 4)) / ATT_WPB, ATT_WPB * 32, 0, stream>>>(
            qkvb, vt, ob, S);
        // out-proj: ob x w_out^T -> f (f32)
        gemm_wmma<512, 512, false, false><<<(256 * 8) / 8, 256, 0, stream>>>(
            ob, w_out + (size_t)l * 512 * 512, out_b + l * 512, f, nullptr, M);
        ln_residual_kernel<<<M, 256, 0, stream>>>(x, xb, f,
                                                  ln1_g + l * 512, ln1_b + l * 512);
        // FFN1 (+ReLU): xb x w_ff1^T -> hbf (bf16)
        gemm_wmma<2048, 512, true, true><<<(256 * 32) / 8, 256, 0, stream>>>(
            xb, w_ff1 + (size_t)l * 2048 * 512, ff1_b + l * 2048, nullptr, hbf, M);
        // FFN2: hbf x w_ff2^T -> f (f32)
        gemm_wmma<512, 2048, false, false><<<(256 * 8) / 8, 256, 0, stream>>>(
            hbf, w_ff2 + (size_t)l * 512 * 2048, ff2_b + l * 512, f, nullptr, M);
        ln_residual_kernel<<<M, 256, 0, stream>>>(x, xb, f,
                                                  ln2_g + l * 512, ln2_b + l * 512);
    }

    decode_kernel<<<M / 8, 256, 0, stream>>>(x, dec_w, dec_b, out, M);
}
